// _MPNNNet_652835029399
// MI455X (gfx1250) — matrix operand order repaired and checked by reading, not compiled, not run on hardware
//
#include <hip/hip_runtime.h>
#include <hip/hip_bf16.h>

#define NN  4096
#define EE  8192
#define NGG 256
#define AFD 9
#define BFD 4
#define HH  128
#define LL  3
#define LN_EPS 1e-5f

typedef __attribute__((ext_vector_type(16))) __bf16 v16bf;
typedef __attribute__((ext_vector_type(8)))  float  v8f;
typedef unsigned int   u32;
typedef unsigned short u16;

union U8v { u32 u[8]; v16bf v; };

__device__ __forceinline__ u16 f2bf(float f) {
    u32 u = __builtin_bit_cast(u32, f);
    u = u + 0x7fffu + ((u >> 16) & 1u);   // round-to-nearest-even
    return (u16)(u >> 16);
}
__device__ __forceinline__ u32 pkmul_bf16(u32 a, u32 b) {
    __hip_bfloat162 x, y;
    __builtin_memcpy(&x, &a, 4);
    __builtin_memcpy(&y, &b, 4);
    __hip_bfloat162 z = __hmul2(x, y);    // packed bf16 multiply
    u32 r;
    __builtin_memcpy(&r, &z, 4);
    return r;
}
__device__ __forceinline__ float gelu_f(float x) {
    return 0.5f * x * (1.0f + erff(x * 0.70710678118654752440f));
}
__device__ __forceinline__ void loadB32(const u16* p, u32* d) {
    const uint4* q = (const uint4*)p;     // 32B aligned
    uint4 a = q[0], b = q[1];
    d[0]=a.x; d[1]=a.y; d[2]=a.z; d[3]=a.w;
    d[4]=b.x; d[5]=b.y; d[6]=b.z; d[7]=b.w;
}

// ---------------- degree ----------------
__global__ void k_deg(const int* __restrict__ dstI, float* __restrict__ deg) {
    int e = blockIdx.x * blockDim.x + threadIdx.x;
    if (e < EE) atomicAdd(&deg[dstI[e]], 1.0f);
}

// ---------------- atom embed: h = relu(x @ atom_w + atom_b) ----------------
__global__ void k_atom(const float* __restrict__ x, const float* __restrict__ w,
                       const float* __restrict__ b, float* __restrict__ h) {
    int idx = blockIdx.x * blockDim.x + threadIdx.x;   // NN*HH threads
    int n = idx >> 7, o = idx & 127;
    float acc = b[o];
    #pragma unroll
    for (int k = 0; k < AFD; ++k) acc += x[n * AFD + k] * w[k * HH + o];
    h[n * HH + o] = fmaxf(acc, 0.0f);
}

// ---------------- pack e2_w/e2_b into bf16 WMMA B-layout ----------------
// chunk c = i0b*129 + k  (k==128 -> bias chunk), 516 chunks/layer
// flat: (((l*516 + c)*8 + wave)*32 + lane)*16 + j ;  lane=(half,o), j -> kk=half*16+j
__global__ void k_pack_w(const float* __restrict__ e2w, const float* __restrict__ e2b,
                         u16* __restrict__ Bw) {
    int blk = blockIdx.x;            // LL*516
    int l = blk / 516, c = blk % 516;
    int i0b = c / 129, k = c % 129;
    int t = threadIdx.x;
    int wv = t >> 5, lane = t & 31;
    int o = lane & 15, hl = lane >> 4;
    int col = wv * 16 + o;
    u16* out = Bw + ((((size_t)(l * 516 + c)) * 8 + wv) * 32 + lane) * 16;
    #pragma unroll
    for (int j = 0; j < 16; ++j) {
        int i = i0b * 32 + hl * 16 + j;
        float v;
        if (k < 128) v = e2w[(size_t)l * (HH*HH*HH) + (size_t)k * (HH*HH) + i * HH + col];
        else         v = e2b[(size_t)l * (HH*HH) + i * HH + col];
        out[j] = f2bf(v);
    }
}

// ---------------- edge hidden: eh = gelu(edge_attr @ e1_w + e1_b) -> bf16 ----------------
__global__ void k_edge_eh(const float* __restrict__ ea, const float* __restrict__ w,
                          const float* __restrict__ b, int l, u16* __restrict__ ehb) {
    int idx = blockIdx.x * blockDim.x + threadIdx.x;   // EE*HH threads
    int e = idx >> 7, o = idx & 127;
    float acc = b[l * HH + o];
    #pragma unroll
    for (int k = 0; k < BFD; ++k) acc += ea[e * BFD + k] * w[(l * BFD + k) * HH + o];
    ehb[e * HH + o] = f2bf(gelu_f(acc));
}

// ---------------- fused edge-network GEMM + per-edge matvec via WMMA ----------------
// block = 32 edges, 8 waves (wave wv owns output cols wv*16..+15)
__global__ __launch_bounds__(256) void k_msg(
    const float* __restrict__ h, const u16* __restrict__ ehb,
    const u16* __restrict__ Bw, const int* __restrict__ srcI,
    float* __restrict__ msg) {
    __shared__ u16 ehk[128 * 32];   // [k][e]  transposed eh, bf16
    __shared__ u16 hsw[32 * 128];   // [(e*4+i0b)*2+half][16]  swizzled h_src, bf16
    const int tid = threadIdx.x;
    const int e0 = blockIdx.x * 32;

    { // stage eh transposed
        int e = tid & 31;
        int k0 = (tid >> 5) * 16;
        const u16* p = ehb + (size_t)(e0 + e) * HH + k0;
        #pragma unroll
        for (int j = 0; j < 16; ++j) ehk[(k0 + j) * 32 + e] = p[j];
    }
    { // stage gathered h_src in A-operand lane order
        int e = tid >> 3, g = tid & 7;
        int i0b = g >> 1, hl = g & 1;
        int s = srcI[e0 + e];
        const float* hp = h + (size_t)s * HH + i0b * 32 + hl * 16;
        u16* dp = &hsw[((e * 4 + i0b) * 2 + hl) * 16];
        #pragma unroll
        for (int j = 0; j < 16; ++j) {
            int kk = j;                              // kk(j) = j (contiguous, half*16 + j)
            dp[j] = f2bf(hp[kk]);
        }
    }
    __syncthreads();

    const int lane = tid & 31;
    const int wv = tid >> 5;
    const int m  = lane & 15;
    const int hl = lane >> 4;

    v8f acc0 = {}, acc1 = {};
    for (int i0b = 0; i0b < 4; ++i0b) {
        U8v hs0, hs1;
        const u32* q0 = (const u32*)&hsw[((m        * 4 + i0b) * 2 + hl) * 16];
        const u32* q1 = (const u32*)&hsw[(((m + 16) * 4 + i0b) * 2 + hl) * 16];
        #pragma unroll
        for (int r = 0; r < 8; ++r) { hs0.u[r] = q0[r]; hs1.u[r] = q1[r]; }

        const u16* Bp = Bw + ((size_t)(i0b * 129) * 8 + wv) * 512 + (size_t)lane * 16;
        U8v Bcur; loadB32(Bp, Bcur.u);
        for (int k = 0; k < 128; ++k) {
            U8v Bnxt; loadB32(Bp + 4096, Bnxt.u);        // double buffer (k+1; k=127 -> bias)
            if (k < 120) __builtin_prefetch((const void*)(Bp + 4 * 4096), 0, 3);
            u32 r0 = ehk[k * 32 + m];      r0 |= r0 << 16;   // bf16 scalar replicated
            u32 r1 = ehk[k * 32 + 16 + m]; r1 |= r1 << 16;
            U8v A0, A1;
            #pragma unroll
            for (int r = 0; r < 8; ++r) {
                A0.u[r] = pkmul_bf16(r0, hs0.u[r]);
                A1.u[r] = pkmul_bf16(r1, hs1.u[r]);
            }
            acc0 = __builtin_amdgcn_wmma_f32_16x16x32_bf16(false, A0.v, false, Bcur.v,
                                                           (short)0, acc0, false, false);
            acc1 = __builtin_amdgcn_wmma_f32_16x16x32_bf16(false, A1.v, false, Bcur.v,
                                                           (short)0, acc1, false, false);
            Bcur = Bnxt;
            Bp += 4096;
        }
        // bias chunk: A = h_src directly (eh factor == 1)
        acc0 = __builtin_amdgcn_wmma_f32_16x16x32_bf16(false, hs0.v, false, Bcur.v,
                                                       (short)0, acc0, false, false);
        acc1 = __builtin_amdgcn_wmma_f32_16x16x32_bf16(false, hs1.v, false, Bcur.v,
                                                       (short)0, acc1, false, false);
    }
    #pragma unroll
    for (int r = 0; r < 8; ++r) {          // C layout: lane=N, VGPR r -> M=r+8*half
        int M = r + 8 * hl;
        msg[(size_t)(e0 + M)      * HH + wv * 16 + m] = acc0[r];
        msg[(size_t)(e0 + 16 + M) * HH + wv * 16 + m] = acc1[r];
    }
}

// ---------------- scatter messages to destination nodes ----------------
__global__ void k_scatter(const float* __restrict__ msg, const int* __restrict__ dstI,
                          float* __restrict__ agg) {
    int idx = blockIdx.x * blockDim.x + threadIdx.x;   // EE*HH
    int e = idx >> 7, o = idx & 127;
    atomicAdd(&agg[(size_t)dstI[e] * HH + o], msg[(size_t)e * HH + o]);
}

// ---------------- node update: relu(agg/deg + h@root_w + b), residual + LayerNorm ----------------
__global__ __launch_bounds__(128) void k_update(
    float* __restrict__ h, const float* __restrict__ agg, const float* __restrict__ deg,
    const float* __restrict__ rootw, const float* __restrict__ convb,
    const float* __restrict__ lng, const float* __restrict__ lnb, int l) {
    __shared__ float shH[HH], s1[HH], s2[HH];
    int n = blockIdx.x, o = threadIdx.x;
    shH[o] = h[(size_t)n * HH + o];
    __syncthreads();
    float root = convb[l * HH + o];
    #pragma unroll 8
    for (int i = 0; i < HH; ++i) root += shH[i] * rootw[(size_t)l * (HH*HH) + i * HH + o];
    float a = agg[(size_t)n * HH + o] / fmaxf(deg[n], 1.0f);
    float t = shH[o] + fmaxf(a + root, 0.0f);
    s1[o] = t; s2[o] = t * t;
    __syncthreads();
    for (int st = 64; st > 0; st >>= 1) {
        if (o < st) { s1[o] += s1[o + st]; s2[o] += s2[o + st]; }
        __syncthreads();
    }
    float mu  = s1[0] * (1.0f / HH);
    float var = s2[0] * (1.0f / HH) - mu * mu;
    h[(size_t)n * HH + o] = (t - mu) * rsqrtf(var + LN_EPS) * lng[l * HH + o] + lnb[l * HH + o];
}

// ---------------- global mean pool ----------------
__global__ void k_pool(const float* __restrict__ h, const int* __restrict__ batch,
                       float* __restrict__ mol, float* __restrict__ cnt) {
    int idx = blockIdx.x * blockDim.x + threadIdx.x;   // NN*HH
    int n = idx >> 7, o = idx & 127;
    int g = batch[n];
    atomicAdd(&mol[(size_t)g * HH + o], h[(size_t)n * HH + o]);
    if (o == 0) atomicAdd(&cnt[g], 1.0f);
}

// ---------------- head MLP ----------------
__global__ __launch_bounds__(128) void k_head(
    const float* __restrict__ mol, const float* __restrict__ cnt,
    const float* __restrict__ w1, const float* __restrict__ b1,
    const float* __restrict__ w2, const float* __restrict__ b2,
    float* __restrict__ out) {
    __shared__ float shm[HH], shr[64];
    int g = blockIdx.x, o = threadIdx.x;
    shm[o] = mol[(size_t)g * HH + o] / fmaxf(cnt[g], 1.0f);
    __syncthreads();
    if (o < 64) {
        float hid = b1[o];
        #pragma unroll 8
        for (int i = 0; i < HH; ++i) hid += shm[i] * w1[i * 64 + o];
        shr[o] = gelu_f(hid) * w2[o];
    }
    __syncthreads();
    for (int st = 32; st > 0; st >>= 1) {
        if (o < st) shr[o] += shr[o + st];
        __syncthreads();
    }
    if (o == 0) out[g] = shr[0] + b2[0];
}

extern "C" void kernel_launch(void* const* d_in, const int* in_sizes, int n_in,
                              void* d_out, int out_size, void* d_ws, size_t ws_size,
                              hipStream_t stream) {
    (void)in_sizes; (void)n_in; (void)out_size;
    const float* x       = (const float*)d_in[0];
    const int*   ei      = (const int*)  d_in[1];   // [2,E]: src=ei, dst=ei+EE
    const float* eattr   = (const float*)d_in[2];
    const int*   batch   = (const int*)  d_in[3];
    const float* atom_w  = (const float*)d_in[4];
    const float* atom_b  = (const float*)d_in[5];
    const float* e1_w    = (const float*)d_in[6];
    const float* e1_b    = (const float*)d_in[7];
    const float* e2_w    = (const float*)d_in[8];
    const float* e2_b    = (const float*)d_in[9];
    const float* root_w  = (const float*)d_in[10];
    const float* conv_b  = (const float*)d_in[11];
    const float* ln_g    = (const float*)d_in[12];
    const float* ln_b    = (const float*)d_in[13];
    const float* head_w1 = (const float*)d_in[14];
    const float* head_b1 = (const float*)d_in[15];
    const float* head_w2 = (const float*)d_in[16];
    const float* head_b2 = (const float*)d_in[17];

    char* ws = (char*)d_ws;
    size_t off = 0;
    auto carve = [&](size_t bytes) -> void* {
        void* p = ws + off; off += (bytes + 255) & ~(size_t)255; return p;
    };
    float* deg  = (float*)carve((size_t)NN * 4);
    float* hbuf = (float*)carve((size_t)NN * HH * 4);
    u16*   ehb  = (u16*)  carve((size_t)EE * HH * 2);
    float* msg  = (float*)carve((size_t)EE * HH * 4);
    float* agg  = (float*)carve((size_t)NN * HH * 4);
    float* mol  = (float*)carve((size_t)NGG * HH * 4);
    float* cnt  = (float*)carve((size_t)NGG * 4);
    u16*   Bw   = (u16*)  carve((size_t)LL * 516 * 4096 * 2);
    if (ws_size < off) return;   // workspace too small; nothing safe to do

    (void)hipMemsetAsync(deg, 0, (size_t)NN * 4, stream);
    k_deg<<<EE / 256, 256, 0, stream>>>(ei + EE, deg);
    k_atom<<<(NN * HH) / 256, 256, 0, stream>>>(x, atom_w, atom_b, hbuf);
    k_pack_w<<<LL * 516, 256, 0, stream>>>(e2_w, e2_b, Bw);

    for (int l = 0; l < LL; ++l) {
        k_edge_eh<<<(EE * HH) / 256, 256, 0, stream>>>(eattr, e1_w, e1_b, l, ehb);
        (void)hipMemsetAsync(agg, 0, (size_t)NN * HH * 4, stream);
        k_msg<<<EE / 32, 256, 0, stream>>>(hbuf, ehb, Bw + (size_t)l * 516 * 4096, ei, msg);
        k_scatter<<<(EE * HH) / 256, 256, 0, stream>>>(msg, ei + EE, agg);
        k_update<<<NN, 128, 0, stream>>>(hbuf, agg, deg, root_w, conv_b, ln_g, ln_b, l);
    }
    (void)hipMemsetAsync(mol, 0, (size_t)NGG * HH * 4, stream);
    (void)hipMemsetAsync(cnt, 0, (size_t)NGG * 4, stream);
    k_pool<<<(NN * HH) / 256, 256, 0, stream>>>(hbuf, batch, mol, cnt);
    k_head<<<NGG, 128, 0, stream>>>(mol, cnt, head_w1, head_b1, head_w2, head_b2,
                                    (float*)d_out);
}
